// SoftmaxAttention_RBF_6957847019895
// MI455X (gfx1250) — matrix operand order repaired and checked by reading, not compiled, not run on hardware
//
#include <hip/hip_runtime.h>

// RBF-kernel attention (no softmax normalization) for MI455X / gfx1250.
// out = exp(Qs Ks^T - diagQ - diagK - 1e9*(1-mask)) @ V,  Qs = Q*mask*64^-0.25.
//
//  * f16 WMMA operands, f32 accumulation (compute-bound: AI ~515 FLOP/B).
//  * All scaling folded into the Q operand: Qop = Q*mask*dn^2*log2e. K is RAW
//    (valid for binary masks: masked keys are annihilated by the -BIG term).
//  * exp2 instead of exp; diag terms pre-seeded into the WMMA accumulator
//    (C-init = -(dq+dk)) so the post-WMMA fixup is exp2 + pack only.
//  * S^T via operand swap: its C-layout IS the A-fragment layout of P.
//  * Cooperative staging of K (f16, fragment-chunk order) and V^T into LDS,
//    DOUBLE-BUFFERED with one barrier/iter: tile kt+1 staged while tile kt computes.

typedef __attribute__((ext_vector_type(16))) _Float16 v16h;
typedef __attribute__((ext_vector_type(8)))  _Float16 v8h;
typedef __attribute__((ext_vector_type(8)))  float    v8f;

#define N_CTX   2048
#define HD      64
#define NHEADS  12
#define WAVES   4
#define M_WAVE  32
#define BLOCK_M (WAVES * M_WAVE)      // 128 rows per workgroup
#define N_TILE  32                    // key tile == WMMA K for f16
#define NT_CNT  (N_CTX / N_TILE)      // 64 key tiles
#define C_QK    0.18033688011112043f  // dn^2 * log2(e)
#define BIG     1.5e9f

// K-fragment chunk offset (in halves): [jt][f][half2][hi][ln] -> 8 contiguous halves
#define KFIDX(jt, f, h2, hi, ln) ((((((((jt)*2 + (f))*2 + (h2))*2 + (hi))*16) + (ln))) * 8)

__global__ __launch_bounds__(128)
void rbf_attn_kernel(const float* __restrict__ Q, const float* __restrict__ K,
                     const float* __restrict__ V, const float* __restrict__ mask,
                     float* __restrict__ Out) {
  __shared__ __align__(16) _Float16 KF[2][2048]; // K tile f16, fragment-chunk order
  __shared__ unsigned VT[2][HD * 17];            // V^T packed f16 pairs, stride 17
  __shared__ float    DK[2][N_TILE];             // NEGATED log2e*diag_K + mask penalty

  const int tid  = threadIdx.x;
  const int lane = tid & 31;
  const int wave = tid >> 5;
  const int hi   = (lane >> 4) & 1;
  const int ln   = lane & 15;

  const int bh    = blockIdx.y;                  // 0..23
  const int bidx  = bh / NHEADS;
  const int qrow0 = blockIdx.x * BLOCK_M + wave * M_WAVE;

  const size_t base = (size_t)bh * N_CTX * HD;
  const float* Qb = Q + base;
  const float* Kb = K + base;
  const float* Vb = V + base;
  const float* mb = mask + (size_t)bidx * N_CTX;

  // ---- Q as B-fragments (lane = row m, k contiguous per half): loop-invariant ----
  v16h  bQ[2][2];
  float ndq[2];
  #pragma unroll
  for (int mt = 0; mt < 2; ++mt) {
    const int   row = qrow0 + mt * 16 + ln;
    const float mv  = mb[row];
    const float sc  = mv * C_QK;            // all normalization on the Q side
    const float* qrow = Qb + (size_t)row * HD + hi * 16;
    float sq = 0.f;
    #pragma unroll
    for (int e = 0; e < 16; ++e) {
      float q0 = qrow[e];
      float q1 = qrow[32 + e];
      sq += q0 * q0 + q1 * q1;
      bQ[mt][0][e] = (_Float16)(q0 * sc);
      bQ[mt][1][e] = (_Float16)(q1 * sc);
    }
    sq += __shfl_xor(sq, 16);
    ndq[mt] = -0.5f * C_QK * mv * mv * sq;  // -log2e * diag_Q[row]
  }

  v8f O[2][4] = {{v8f{}, v8f{}, v8f{}, v8f{}}, {v8f{}, v8f{}, v8f{}, v8f{}}};

  // staging roles (constant per thread)
  const int sj  = tid >> 2;        // K staging: row 0..31
  const int skb = tid & 3;         // K staging: 16-wide k block 0..3
  const int vc  = tid & 63;        // V staging: output column
  const int vp0 = tid >> 6;        // V staging: first pair 0..1

  // ---- cooperative staging of one key tile into buffer `buf` ----
  auto stage = [&](int buf, int j0) {
    {   // K tile -> f16 LDS in fragment-chunk order, plus negated diag_K
      const float4* kr = (const float4*)(Kb + (size_t)(j0 + sj) * HD + skb * 16);
      float4 k0 = kr[0], k1 = kr[1], k2 = kr[2], k3 = kr[3];
      float sk = k0.x*k0.x + k0.y*k0.y + k0.z*k0.z + k0.w*k0.w
               + k1.x*k1.x + k1.y*k1.y + k1.z*k1.z + k1.w*k1.w
               + k2.x*k2.x + k2.y*k2.y + k2.z*k2.z + k2.w*k2.w
               + k3.x*k3.x + k3.y*k3.y + k3.z*k3.z + k3.w*k3.w;
      v8h lo, hi8;
      lo[0]=(_Float16)k0.x; lo[1]=(_Float16)k0.y; lo[2]=(_Float16)k0.z; lo[3]=(_Float16)k0.w;
      lo[4]=(_Float16)k1.x; lo[5]=(_Float16)k1.y; lo[6]=(_Float16)k1.z; lo[7]=(_Float16)k1.w;
      hi8[0]=(_Float16)k2.x; hi8[1]=(_Float16)k2.y; hi8[2]=(_Float16)k2.z; hi8[3]=(_Float16)k2.w;
      hi8[4]=(_Float16)k3.x; hi8[5]=(_Float16)k3.y; hi8[6]=(_Float16)k3.z; hi8[7]=(_Float16)k3.w;
      const int jt = sj >> 4, lnj = sj & 15, f = skb >> 1, h2 = skb & 1;
      *(v8h*)&KF[buf][KFIDX(jt, f, 0, h2, lnj)] = lo;
      *(v8h*)&KF[buf][KFIDX(jt, f, 1, h2, lnj)] = hi8;
      sk += __shfl_xor(sk, 1);     // the 4 threads of a row are adjacent lanes
      sk += __shfl_xor(sk, 2);
      if (skb == 0)
        DK[buf][sj] = -(0.5f * C_QK * sk + BIG * (1.f - mb[j0 + sj]));
    }
    // V^T tile -> LDS (packed f16 pairs)
    #pragma unroll
    for (int r = 0; r < 8; ++r) {
      const int p = vp0 + 2 * r;   // pair index 0..15 (rows j = 2p, 2p+1)
      float v0 = Vb[(size_t)(j0 + 2 * p)     * HD + vc];
      float v1 = Vb[(size_t)(j0 + 2 * p + 1) * HD + vc];
      union { _Float16 h[2]; unsigned u; } pk;
      pk.h[0] = (_Float16)v0; pk.h[1] = (_Float16)v1;
      VT[buf][vc * 17 + p] = pk.u;
    }
    if (j0 + N_TILE < N_CTX) {     // uniform; hint the tile after this one
      __builtin_prefetch(Kb + (size_t)(j0 + N_TILE) * HD, 0, 1);
      __builtin_prefetch(Vb + (size_t)(j0 + N_TILE) * HD, 0, 1);
    }
  };

  stage(0, 0);   // prologue: tile 0 into buffer 0

  for (int kt = 0; kt < NT_CNT; ++kt) {
    const int cur = kt & 1;
    __syncthreads();               // buffer `cur` staged; buffer 1-cur free to overwrite

    if (kt + 1 < NT_CNT)           // overlap: stage next tile while computing this one
      stage(1 - cur, (kt + 1) * N_TILE);

    // ---- per-wave fragment loads from LDS buffer `cur` ----
    union { v8h c[2]; v16h v; } aK[2][2];
    #pragma unroll
    for (int jt = 0; jt < 2; ++jt)
      #pragma unroll
      for (int f = 0; f < 2; ++f) {
        aK[jt][f].c[0] = *(const v8h*)&KF[cur][KFIDX(jt, f, 0, hi, ln)];
        aK[jt][f].c[1] = *(const v8h*)&KF[cur][KFIDX(jt, f, 1, hi, ln)];
      }

    float nk0[8], nk1[8];          // negated diag_K for this lane's rows (merged ds loads)
    #pragma unroll
    for (int v = 0; v < 8; ++v) {
      nk0[v] = DK[cur][8 * hi + v];
      nk1[v] = DK[cur][16 + 8 * hi + v];
    }

    union { unsigned u[8]; v16h v; } bV[4];
    const unsigned* vtp = &VT[cur][ln * 17 + 4 * hi];
    #pragma unroll
    for (int ct = 0; ct < 4; ++ct)
      #pragma unroll
      for (int q = 0; q < 8; ++q)
        bV[ct].u[q] = vtp[ct * 272 + q + (q & 4)];     // constant offsets -> ds immediates

    #pragma unroll
    for (int mt = 0; mt < 2; ++mt) {
      // accumulator pre-seeded with -(dq + dk): post-WMMA fixup is exp2+pack only
      v8f Ct0, Ct1;
      #pragma unroll
      for (int v = 0; v < 8; ++v) { Ct0[v] = ndq[mt] + nk0[v]; Ct1[v] = ndq[mt] + nk1[v]; }

      // S^T - dq - dk = K * Qop^T + Cinit : C layout == A-fragment layout of P
      Ct0 = __builtin_amdgcn_wmma_f32_16x16x32_f16(false, aK[0][0].v, false, bQ[mt][0], (short)0, Ct0, false, false);
      Ct0 = __builtin_amdgcn_wmma_f32_16x16x32_f16(false, aK[0][1].v, false, bQ[mt][1], (short)0, Ct0, false, false);
      Ct1 = __builtin_amdgcn_wmma_f32_16x16x32_f16(false, aK[1][0].v, false, bQ[mt][0], (short)0, Ct1, false, false);
      Ct1 = __builtin_amdgcn_wmma_f32_16x16x32_f16(false, aK[1][1].v, false, bQ[mt][1], (short)0, Ct1, false, false);

      v16h aP;
      #pragma unroll
      for (int v = 0; v < 8; ++v) {
        aP[v]     = (_Float16)__builtin_amdgcn_exp2f(Ct0[v]);
        aP[8 + v] = (_Float16)__builtin_amdgcn_exp2f(Ct1[v]);
      }

      #pragma unroll
      for (int ct = 0; ct < 4; ++ct)
        O[mt][ct] = __builtin_amdgcn_wmma_f32_16x16x32_f16(false, aP, false, bV[ct].v, (short)0, O[mt][ct], false, false);
    }
  }

  // ---- store: C layout element (row = v + 8hi, col = ct*16 + ln) ----
  #pragma unroll
  for (int mt = 0; mt < 2; ++mt) {
    float* ob = Out + base + (size_t)(qrow0 + mt * 16) * HD;
    #pragma unroll
    for (int ct = 0; ct < 4; ++ct) {
      #pragma unroll
      for (int v = 0; v < 8; ++v)
        ob[(size_t)(v + 8 * hi) * HD + ct * 16 + ln] = O[mt][ct][v];
    }
  }
}

extern "C" void kernel_launch(void* const* d_in, const int* in_sizes, int n_in,
                              void* d_out, int out_size, void* d_ws, size_t ws_size,
                              hipStream_t stream) {
  (void)in_sizes; (void)n_in; (void)out_size; (void)d_ws; (void)ws_size;
  const float* Q    = (const float*)d_in[0];
  const float* K    = (const float*)d_in[1];
  const float* V    = (const float*)d_in[2];
  const float* mask = (const float*)d_in[3];
  float* Out = (float*)d_out;

  dim3 grid(N_CTX / BLOCK_M, 2 * NHEADS);   // 16 x 24
  dim3 block(WAVES * 32);                    // 4 wave32s
  rbf_attn_kernel<<<grid, block, 0, stream>>>(Q, K, V, mask, Out);
}
